// Model_996432413561
// MI455X (gfx1250) — matrix operand order repaired and checked by reading, not compiled, not run on hardware
//
#include <hip/hip_runtime.h>
#include <hip/hip_bf16.h>

#define N_ATOMS   512
#define N_SAMPLES 32768
#define N_FREQ    16385
#define L_REAL    32770          // 2 * 16385
#define N_BATCH   64
#define N_ITER    64
#define KP        33280          // padded K = 16 chunks * 65 steps * 32
#define KCHUNKS   16
#define STEPS_PER_CHUNK 65       // KP / (KCHUNKS*32)

typedef __attribute__((ext_vector_type(16))) __bf16        v16bf;
typedef __attribute__((ext_vector_type(8)))  float         v8f;
typedef __attribute__((ext_vector_type(4)))  unsigned int  u32x4;

static __device__ __forceinline__ unsigned short f32_to_bf16(float f) {
    unsigned int u = __float_as_uint(f);
    u = (u + 0x7FFFu + ((u >> 16) & 1u)) >> 16;   // round-to-nearest-even
    return (unsigned short)u;
}

// ---------------------------------------------------------------- norms ----
__global__ void norm_kernel(const float* __restrict__ atoms, float* __restrict__ invn) {
    int a = blockIdx.x;
    const float* row = atoms + (size_t)a * L_REAL;
    float s = 0.f;
    for (int l = threadIdx.x; l < L_REAL; l += blockDim.x) { float v = row[l]; s += v * v; }
    __shared__ float sh[256];
    sh[threadIdx.x] = s; __syncthreads();
    for (int off = 128; off > 0; off >>= 1) {
        if (threadIdx.x < off) sh[threadIdx.x] += sh[threadIdx.x + off];
        __syncthreads();
    }
    if (threadIdx.x == 0) invn[a] = 1.0f / sqrtf(sh[0]);
}

// ------------------------------------------------------- bf16 conversion ---
__global__ void cvt_atoms_kernel(const float* __restrict__ atoms,
                                 const float* __restrict__ invn,
                                 unsigned short* __restrict__ anb) {
    size_t t = (size_t)blockIdx.x * blockDim.x + threadIdx.x;
    if (t >= (size_t)N_ATOMS * KP) return;
    int a = (int)(t / KP);
    int l = (int)(t - (size_t)a * KP);
    float v = (l < L_REAL) ? atoms[(size_t)a * L_REAL + l] * invn[a] : 0.0f;
    anb[t] = f32_to_bf16(v);
}

__global__ void cvt_res_kernel(const float* __restrict__ res,
                               unsigned short* __restrict__ resb) {
    size_t t = (size_t)blockIdx.x * blockDim.x + threadIdx.x;
    if (t >= (size_t)N_BATCH * KP) return;
    int b = (int)(t / KP);
    int l = (int)(t - (size_t)b * KP);
    float v = (l < L_REAL) ? res[(size_t)b * L_REAL + l] : 0.0f;
    resb[t] = f32_to_bf16(v);
}

__global__ void zero_kernel(float* __restrict__ p, int n) {
    int t = blockIdx.x * blockDim.x + threadIdx.x;
    if (t < n) p[t] = 0.0f;
}

// ---------------------------------------------------- WMMA split-K GEMM ----
// dots_partial[kc][b][a] = sum over K-chunk kc of res[b][k] * an[a][k]
// Each wave computes a 16(M) x 32(N) strip: one A load feeds two WMMAs.
__global__ void gemm_dots_kernel(const unsigned short* __restrict__ resb,  // [64][KP]
                                 const unsigned short* __restrict__ anb,   // [512][KP]
                                 float* __restrict__ part) {               // [16][64][512]
    const int lane = threadIdx.x & 31;
    const int wave = threadIdx.x >> 5;                     // 0..7
    const int mt   = blockIdx.x;                           // 0..3  (batch tiles)
    const int ntp  = (blockIdx.y * 8 + wave) * 2;          // 0..30 (atom tile pair)
    const int kc   = blockIdx.z;                           // 0..15 (split-K)
    const int r    = lane & 15;
    const int hi   = lane >> 4;

    const unsigned short* arow  = resb + (size_t)(mt * 16 + r) * KP;        // A row = batch
    const unsigned short* brow0 = anb  + (size_t)(ntp * 16 + r) * KP;       // B col = atom
    const unsigned short* brow1 = anb  + (size_t)((ntp + 1) * 16 + r) * KP;

    int k0 = kc * STEPS_PER_CHUNK * 32;
    v8f acc0 = {}, acc1 = {};
    for (int s = 0; s < STEPS_PER_CHUNK; ++s, k0 += 32) {
        union { v16bf v; u32x4 u[2]; } A, B0, B1;
        // A 16x32 bf16: elems 0-15 -> K = k0+hi*16.. (contiguous)
        A.u[0] = *(const u32x4*)(arow + k0 + hi * 16);
        A.u[1] = *(const u32x4*)(arow + k0 + hi * 16 + 8);
        // B 32x16 bf16: elems 0-15 -> K = k0+hi*16.. (contiguous along atom row)
        const u32x4* bp0 = (const u32x4*)(brow0 + k0 + hi * 16);
        B0.u[0] = bp0[0];
        B0.u[1] = bp0[1];
        const u32x4* bp1 = (const u32x4*)(brow1 + k0 + hi * 16);
        B1.u[0] = bp1[0];
        B1.u[1] = bp1[1];
        acc0 = __builtin_amdgcn_wmma_f32_16x16x32_bf16(false, A.v, false, B0.v,
                                                       (short)0, acc0, false, false);
        acc1 = __builtin_amdgcn_wmma_f32_16x16x32_bf16(false, A.v, false, B1.v,
                                                       (short)0, acc1, false, false);
    }
    // C layout: element v -> (M = v + 8*hi, N = lane&15)
    const int n0    = ntp * 16 + (lane & 15);
    const int mbase = mt * 16 + hi * 8;
    float* dst = part + (size_t)kc * N_BATCH * N_ATOMS;
    #pragma unroll
    for (int v = 0; v < 8; ++v) {
        dst[(size_t)(mbase + v) * N_ATOMS + n0]      = acc0[v];
        dst[(size_t)(mbase + v) * N_ATOMS + n0 + 16] = acc1[v];
    }
}

// ----------------------------------------------- split-K reduce + argmax ---
__global__ void argmax_kernel(const float* __restrict__ part, int* __restrict__ idx) {
    int b = blockIdx.x;
    int a = threadIdx.x;                              // 512 threads = 512 atoms
    float s = 0.f;
    for (int kc = 0; kc < KCHUNKS; ++kc)
        s += part[((size_t)kc * N_BATCH + b) * N_ATOMS + a];
    __shared__ float sval[512];
    __shared__ int   sidx[512];
    sval[a] = s; sidx[a] = a; __syncthreads();
    for (int off = 256; off > 0; off >>= 1) {
        if (a < off) {
            float v2 = sval[a + off]; int i2 = sidx[a + off];
            if (v2 > sval[a] || (v2 == sval[a] && i2 < sidx[a])) { sval[a] = v2; sidx[a] = i2; }
        }
        __syncthreads();
    }
    if (a == 0) idx[b] = sidx[0];
}

// ------------------------------------------- residual update (fused cvt) ---
__global__ void update_kernel(const float* __restrict__ atoms,
                              const float* __restrict__ invn,
                              const int* __restrict__ idx,
                              float* __restrict__ res,
                              float* __restrict__ rec,
                              unsigned short* __restrict__ resb) {
    int b = blockIdx.x;
    int l = blockIdx.y * blockDim.x + threadIdx.x;
    if (l >= L_REAL) return;
    int a = idx[b];
    float an   = atoms[(size_t)a * L_REAL + l] * invn[a];
    size_t o   = (size_t)b * L_REAL + l;
    float r    = res[o];
    float best = an * r;
    float nr   = r - best;
    res[o] = nr;
    rec[o] += best;
    resb[(size_t)b * KP + l] = f32_to_bf16(nr);
}

// ------------------------------------------------------------- FFT core ----
// in-place radix-2 DIT complex FFT over LDS; sign = -1 forward, +1 inverse
static __device__ void cfft_lds(float2* s, int n, int logn, float sign) {
    const int tid = threadIdx.x, nt = blockDim.x;
    for (int i = tid; i < n; i += nt) {
        int j = (int)(__brev((unsigned)i) >> (32 - logn));
        if (j > i) { float2 t = s[i]; s[i] = s[j]; s[j] = t; }
    }
    __syncthreads();
    const float PI = 3.14159265358979323846f;
    for (int st = 1; st <= logn; ++st) {
        const int half = 1 << (st - 1);
        for (int t = tid; t < (n >> 1); t += nt) {
            int j = t & (half - 1);
            int base = ((t >> (st - 1)) << st) + j;
            float ang = sign * PI * (float)j / (float)half;   // sign*2*pi*j/m
            float sn, c; sincosf(ang, &sn, &c);
            float2 u = s[base];
            float2 w = s[base + half];
            float tr = c * w.x - sn * w.y;
            float ti = c * w.y + sn * w.x;
            s[base]        = make_float2(u.x + tr, u.y + ti);
            s[base + half] = make_float2(u.x - tr, u.y - ti);
        }
        __syncthreads();
    }
}

// rfft(32768 real) via 16384-pt complex FFT + unpack; out = [Re(0..16384)|Im(0..16384)]
__global__ void rfft_kernel(const float* __restrict__ x, float* __restrict__ res) {
    extern __shared__ float2 smem[];
    const int b = blockIdx.x;
    const float* xr = x + (size_t)b * N_SAMPLES;
    for (int i = threadIdx.x; i < 16384; i += blockDim.x)
        smem[i] = make_float2(xr[2 * i], xr[2 * i + 1]);
    __syncthreads();
    cfft_lds(smem, 16384, 14, -1.0f);
    float* out = res + (size_t)b * L_REAL;
    const float PI = 3.14159265358979323846f;
    for (int k = threadIdx.x; k <= 16384; k += blockDim.x) {
        float2 Zk = smem[k & 16383];
        float2 Zm = smem[(16384 - k) & 16383];
        float Ex = 0.5f * (Zk.x + Zm.x), Ey = 0.5f * (Zk.y - Zm.y);
        float Dx = 0.5f * (Zk.x - Zm.x), Dy = 0.5f * (Zk.y + Zm.y);
        float Ox = Dy, Oy = -Dx;                       // O = -i * D
        float ang = -PI * (float)k / 16384.0f;         // -2*pi*k/32768
        float sn, c; sincosf(ang, &sn, &c);
        out[k]             = Ex + c * Ox - sn * Oy;
        out[N_FREQ + k]    = Ey + c * Oy + sn * Ox;
    }
}

// irfft: spectrum [Re|Im] -> 32768 real samples, via packed 16384-pt inverse FFT
__global__ void irfft_kernel(const float* __restrict__ rec, float* __restrict__ out) {
    extern __shared__ float2 smem[];
    const int b = blockIdx.x;
    const float* Xr = rec + (size_t)b * L_REAL;
    const float* Xi = Xr + N_FREQ;
    const float PI = 3.14159265358979323846f;
    for (int k = threadIdx.x; k < 16384; k += blockDim.x) {
        int mk = 16384 - k;
        float Ex = 0.5f * (Xr[k] + Xr[mk]), Ey = 0.5f * (Xi[k] - Xi[mk]);
        float Dx = 0.5f * (Xr[k] - Xr[mk]), Dy = 0.5f * (Xi[k] + Xi[mk]);
        float ang = PI * (float)k / 16384.0f;          // +2*pi*k/32768
        float sn, c; sincosf(ang, &sn, &c);
        float Ox = c * Dx - sn * Dy, Oy = c * Dy + sn * Dx;
        smem[k] = make_float2(Ex - Oy, Ey + Ox);       // Z = E + i*O
    }
    __syncthreads();
    cfft_lds(smem, 16384, 14, 1.0f);
    float* o = out + (size_t)b * N_SAMPLES;
    const float inv = 1.0f / 16384.0f;
    for (int i = threadIdx.x; i < 16384; i += blockDim.x) {
        o[2 * i]     = smem[i].x * inv;
        o[2 * i + 1] = smem[i].y * inv;
    }
}

// ----------------------------------------------------------------- host ----
extern "C" void kernel_launch(void* const* d_in, const int* in_sizes, int n_in,
                              void* d_out, int out_size, void* d_ws, size_t ws_size,
                              hipStream_t stream) {
    const float* x     = (const float*)d_in[0];   // (64,1,32768)
    const float* atoms = (const float*)d_in[1];   // (1,512,32770)

    float* recon = (float*)d_out;                               // 64*32768
    float* res   = recon + (size_t)N_BATCH * N_SAMPLES;         // 64*32770 (2nd output)

    char*  ws  = (char*)d_ws;
    size_t off = 0;
    auto alloc = [&](size_t bytes) {
        char* p = ws + off;
        off += (bytes + 255) & ~(size_t)255;
        return (void*)p;
    };
    float*          invn = (float*)         alloc((size_t)N_ATOMS * 4);
    int*            idx  = (int*)           alloc((size_t)N_BATCH * 4);
    unsigned short* anb  = (unsigned short*)alloc((size_t)N_ATOMS * KP * 2);
    unsigned short* resb = (unsigned short*)alloc((size_t)N_BATCH * KP * 2);
    float*          rec  = (float*)         alloc((size_t)N_BATCH * L_REAL * 4);
    float*          part = (float*)         alloc((size_t)KCHUNKS * N_BATCH * N_ATOMS * 4);

    // Setup
    norm_kernel<<<N_ATOMS, 256, 0, stream>>>(atoms, invn);
    {
        size_t tot = (size_t)N_ATOMS * KP;
        cvt_atoms_kernel<<<(unsigned)((tot + 255) / 256), 256, 0, stream>>>(atoms, invn, anb);
    }
    rfft_kernel<<<N_BATCH, 512, 16384 * sizeof(float2), stream>>>(x, res);
    zero_kernel<<<(N_BATCH * L_REAL + 255) / 256, 256, 0, stream>>>(rec, N_BATCH * L_REAL);
    {
        size_t tot = (size_t)N_BATCH * KP;
        cvt_res_kernel<<<(unsigned)((tot + 255) / 256), 256, 0, stream>>>(res, resb);
    }

    // Matching-pursuit iterations
    for (int it = 0; it < N_ITER; ++it) {
        gemm_dots_kernel<<<dim3(4, 2, KCHUNKS), 256, 0, stream>>>(resb, anb, part);
        argmax_kernel<<<N_BATCH, 512, 0, stream>>>(part, idx);
        update_kernel<<<dim3(N_BATCH, (L_REAL + 255) / 256), 256, 0, stream>>>(
            atoms, invn, idx, res, rec, resb);
    }

    // Reconstruction
    irfft_kernel<<<N_BATCH, 512, 16384 * sizeof(float2), stream>>>(rec, recon);
}
